// MLPEMABlock_65463891525760
// MI455X (gfx1250) — matrix operand order repaired and checked by reading, not compiled, not run on hardware
//
#include <hip/hip_runtime.h>
#include <hip/hip_bf16.h>

// ---------------------------------------------------------------------------
// MLPEMABlock for MI455X (gfx1250, wave32).
//   K1: windowed-geometric EMA var via O(1)/elem sliding-window recurrence.
//   K2: TDM (tensor_load_to_lds) stages the contiguous var tile into LDS,
//       RMS-norm (÷192), then (B*S,64)@(64,128) on v_wmma_f32_16x16x32_f16
//       with one-time f32->f16 staging, and a coalescing LDS transpose store.
// ---------------------------------------------------------------------------

#define Bsz   8
#define Fin   64
#define Slen  8192
#define KWIN  201
#define DCOMB 192
#define NOUT  128      // 2*OUT
#define OUTC  64
#define EPSF  1e-8f

#define CHUNK 128
#define NCH   (Slen / CHUNK)   // 64 chunks per (b,f) row
#define MTILE 128              // time rows per block in kernel 2

typedef __attribute__((ext_vector_type(16))) _Float16 v16h;
typedef __attribute__((ext_vector_type(8)))  float    v8f;
typedef unsigned int u32x4 __attribute__((ext_vector_type(4)));
typedef int          i32x4 __attribute__((ext_vector_type(4)));
typedef int          i32x8 __attribute__((ext_vector_type(8)));

#if defined(__has_builtin)
#if __has_builtin(__builtin_amdgcn_tensor_load_to_lds) && __has_builtin(__builtin_amdgcn_s_wait_tensorcnt)
#define HAVE_TDM 1
#endif
#endif

// ---------------- Kernel 1: ema_var -> ws as (b, t, f) row-major ------------
__global__ __launch_bounds__(256) void ema_var_kernel(
    const float* __restrict__ x,            // (B, F, S)
    const float* __restrict__ log_hl_var,   // (F)
    float* __restrict__ var_out)            // (B*S, 64)
{
    const int tid   = blockIdx.x * blockDim.x + threadIdx.x;
    const int f     = tid & (Fin - 1);          // lane-fastest -> coalesced ws stores
    const int chunk = (tid >> 6) & (NCH - 1);
    const int b     = tid >> 12;

    const float hl      = __expf(log_hl_var[f]);
    const float l2a     = -1.0f / hl;           // log2(alpha) = -1/halflife
    const float a       = exp2f(l2a);
    const float a201    = exp2f(201.0f * l2a);
    const float inv1ma  = 1.0f / (1.0f - a);
    const float denFull = (1.0f - a201) * inv1ma;

    const float* __restrict__ xr = x + ((size_t)b * Fin + f) * Slen;
    const int t0 = chunk * CHUNK;

    // Direct init of the 201-tap window sums at t = t0 (zero-padded history).
    float s1 = 0.0f, s2 = 0.0f, ap = 1.0f;
    const int kmax = (t0 < KWIN - 1) ? t0 : (KWIN - 1);
    for (int k = 0; k <= kmax; ++k) {
        const float v = xr[t0 - k];
        s1 = fmaf(ap, v, s1);
        s2 = fmaf(ap * v, v, s2);
        ap *= a;
    }

    // Sliding-window recurrence: s[t+1] = a*s[t] + x[t+1] - a^201 * x[t-200]
    for (int t = t0; t < t0 + CHUNK; ++t) {
        if ((t & 63) == 0) __builtin_prefetch(xr + t + 256, 0, 0);  // global_prefetch_b8
        const float den = (t >= KWIN - 1)
                        ? denFull
                        : (1.0f - exp2f(l2a * (float)(t + 1))) * inv1ma;
        const float inv = 1.0f / (den + EPSF);
        const float m1  = s1 * inv;
        const float var = fmaf(-m1, m1, s2 * inv);
        var_out[((size_t)b * Slen + t) * Fin + f] = var;

        const int   tn = t + 1;
        const float xn = (tn < Slen) ? xr[tn] : 0.0f;
        const int   to = tn - KWIN;
        const float xo = (to >= 0) ? xr[to] : 0.0f;
        s1 = fmaf(a, s1, fmaf(-a201, xo, xn));
        s2 = fmaf(a, s2, fmaf(-a201, xo * xo, xn * xn));
    }
}

// ---------------- Kernel 2: TDM stage + RMS-norm + WMMA linear --------------
__global__ __launch_bounds__(256) void norm_linear_wmma_kernel(
    const float* __restrict__ var_in,   // (B*S, 64)
    const float* __restrict__ norm_w,   // (192) -- use [128..191]
    const float* __restrict__ lin_W,    // (192, 128) -- use rows [128..191]
    const float* __restrict__ lin_b,    // (128)
    float* __restrict__ out)            // [output (B,64,S) | skip (B,64,S)]
{
    // Overlay region: [f32 var staging 8192 | f16 cn 128x80] then outT 128x130 f32.
    __shared__ float    smemOver[MTILE * 130];
    __shared__ _Float16 sWh[NOUT][80];          // W^T as f16: [n][k], 16B-aligned rows
    __shared__ float    lb[NOUT];
    __shared__ float    nw[64];

    float*    __restrict__ sVar = smemOver;                        // 8192 f32 (TDM dest)
    _Float16* __restrict__ cnh  = (_Float16*)(smemOver + 8192);    // 128x80 halves
    float*    __restrict__ oT   = smemOver;                        // pitch 130 (reused)

    const int blocksPerB = Slen / MTILE;        // 64
    const int b   = blockIdx.x / blocksPerB;
    const int t0  = (blockIdx.x % blocksPerB) * MTILE;
    const int tid = threadIdx.x;

    // ---- Stage the contiguous var tile (8192 f32 = 32 KB) into LDS. -------
    const float* tileSrc = var_in + ((size_t)b * Slen + t0) * Fin;
#if defined(HAVE_TDM)
    if (tid == 0) {
        const unsigned ldsOff = (unsigned)(unsigned long long)(uintptr_t)sVar;
        const unsigned long long ga = (unsigned long long)(uintptr_t)tileSrc;
        u32x4 g0;                               // D# group 0 (ISA 8.3)
        g0[0] = 1u;                             // count=1, user mode, no gather
        g0[1] = ldsOff;                         // lds_addr
        g0[2] = (unsigned)(ga & 0xFFFFFFFFull); // global_addr[31:0]
        g0[3] = (unsigned)((ga >> 32) & 0x1FFFFFFull) | (2u << 30); // addr[56:32] | type=2
        i32x8 g1;                               // D# group 1 (ISA 8.4), 1-D copy
        g1[0] = (int)(2u << 16);                // workgroup_mask=0, data_size=2 (4B)
        g1[1] = (int)((8192u & 0xFFFFu) << 16); // tensor_dim0[15:0] << 16
        g1[2] = (int)((8192u >> 16) | (1u << 16)); // tensor_dim0[31:16] | tensor_dim1=1
        g1[3] = (int)(8192u << 16);             // tile_dim0 = 8192
        g1[4] = 1;                              // tile_dim1 = 1
        g1[5] = 8192;                           // tensor_dim0_stride (unused, degenerate)
        g1[6] = 0;
        g1[7] = 0;
        i32x4 gz = {0, 0, 0, 0};
#if defined(__clang_major__) && (__clang_major__ >= 23)
        i32x8 gz8 = {0, 0, 0, 0, 0, 0, 0, 0};
        __builtin_amdgcn_tensor_load_to_lds(g0, g1, gz, gz, gz8, 0);
#else
        __builtin_amdgcn_tensor_load_to_lds(g0, g1, gz, gz, 0);
#endif
        __builtin_amdgcn_s_wait_tensorcnt(0);
    }
#else
    for (int i = tid; i < MTILE * Fin; i += 256) sVar[i] = tileSrc[i];
#endif

    // ---- Stage W rows 128..191 transposed to f16 [n][k]; bias; norm_w. ----
    for (int i = tid; i < 64 * NOUT; i += 256) {
        const int k = i >> 7, n = i & 127;      // coalesced global read along n
        sWh[n][k] = (_Float16)lin_W[(size_t)(128 + k) * NOUT + n];
    }
    if (tid < NOUT) lb[tid] = lin_b[tid];
    if (tid < 64)   nw[tid] = norm_w[128 + tid];
    __syncthreads();

    // ---- RMS over 192-wide combined vector (128 channels exactly zero). ---
    if (tid < MTILE) {
        float sum = 0.0f;
        for (int f = 0; f < 64; ++f) { const float v = sVar[tid * 64 + f]; sum = fmaf(v, v, sum); }
        const float scale = rsqrtf(sum * (1.0f / (float)DCOMB) + EPSF);
        for (int f = 0; f < 64; ++f)
            cnh[tid * 80 + f] = (_Float16)(sVar[tid * 64 + f] * scale * nw[f]);
    }
    __syncthreads();

    // ---- WMMA: (16t x 64f) @ (64f x 128n) per wave, f16 in, f32 accum. ----
    const int wave = tid >> 5;                  // 0..7 -> 16-row slab
    const int lane = tid & 31;
    const int m0   = wave * 16;
    const int mrow = m0 + (lane & 15);
    const int hiA  = (lane >> 4) ? 8 : 0;       // A: lanes 16-31 hold K+8 block
    const int hiB  = (lane >> 4) ? 16 : 0;      // unused: B is read with hiA, same K order as A
    const int ncol = lane & 15;

    v8f acc[8];
#pragma unroll
    for (int nt = 0; nt < 8; ++nt) {            // bias folded into C init
        const float bv = lb[nt * 16 + ncol];
        v8f t;
#pragma unroll
        for (int j = 0; j < 8; ++j) t[j] = bv;
        acc[nt] = t;
    }

#pragma unroll
    for (int kc = 0; kc < 64; kc += 32) {
        v16h A;                                 // ISA 16-bit A 16x32 layout
#pragma unroll
        for (int j = 0; j < 8; ++j) {
            A[j]     = cnh[mrow * 80 + kc + hiA + j];
            A[8 + j] = cnh[mrow * 80 + kc + 16 + hiA + j];
        }
#pragma unroll
        for (int nt = 0; nt < 8; ++nt) {
            v16h Bm;                            // B 32x16: same K order as A (two 8-half runs)
#pragma unroll
            for (int j = 0; j < 8; ++j) {
                Bm[j]     = sWh[nt * 16 + ncol][kc + hiA + j];
                Bm[8 + j] = sWh[nt * 16 + ncol][kc + 16 + hiA + j];
            }
            acc[nt] = __builtin_amdgcn_wmma_f32_16x16x32_f16(
                false, A, false, Bm, (short)0, acc[nt], false, false);
        }
    }
    __syncthreads();   // all waves done reading cnh/sVar -> safe to overlay oT

    // ---- Transpose through LDS so global stores are contiguous along t. ---
#pragma unroll
    for (int nt = 0; nt < 8; ++nt) {
        const int n = nt * 16 + ncol;
#pragma unroll
        for (int r = 0; r < 8; ++r) {           // C layout: lanes16-31 are rows M+8
            const int row = m0 + ((lane < 16) ? r : (r + 8));
            oT[n * 130 + row] = acc[nt][r];
        }
    }
    __syncthreads();

    for (int i = tid; i < NOUT * MTILE; i += 256) {
        const int n  = i >> 7;                  // output channel 0..127
        const int tt = i & 127;                 // contiguous across lanes -> coalesced
        const size_t base = (n >= OUTC)
            ? ((size_t)(b * OUTC + (n - OUTC)) * Slen)                          // output
            : ((size_t)Bsz * OUTC * Slen + (size_t)(b * OUTC + n) * Slen);      // skip
        out[base + t0 + tt] = oT[n * 130 + tt];
    }
}

// ---------------------------------------------------------------------------
extern "C" void kernel_launch(void* const* d_in, const int* in_sizes, int n_in,
                              void* d_out, int out_size, void* d_ws, size_t ws_size,
                              hipStream_t stream) {
    (void)in_sizes; (void)n_in; (void)out_size; (void)ws_size;
    const float* x          = (const float*)d_in[0];
    const float* log_hl_var = (const float*)d_in[3];
    const float* norm_w     = (const float*)d_in[8];
    const float* lin_W      = (const float*)d_in[9];
    const float* lin_b      = (const float*)d_in[10];
    float* out    = (float*)d_out;
    float* var_ws = (float*)d_ws;               // B*S*64 floats = 16 MiB

    const int k1_threads = Bsz * Fin * NCH;     // 32768
    ema_var_kernel<<<k1_threads / 256, 256, 0, stream>>>(x, log_hl_var, var_ws);

    norm_linear_wmma_kernel<<<Bsz * (Slen / MTILE), 256, 0, stream>>>(
        var_ws, norm_w, lin_W, lin_b, out);
}
